// SelfAttention_30064771072473
// MI455X (gfx1250) — matrix operand order repaired and checked by reading, not compiled, not run on hardware
//
#include <hip/hip_runtime.h>
#include <math.h>

// ---------------------------------------------------------------------------
// CDNA5 (gfx1250) dual-branch feature-attention block.
// All GEMMs: bf16 operands (k-major), V_WMMA_F32_16X16X32_BF16, f32 accum.
// 256x128 block tile, 8 waves (4x2), 64x64 wave tile (16 WMMA / 16 ds_b128),
// double-buffered LDS staging, conversions hoisted out of the GEMM entirely.
// ---------------------------------------------------------------------------

typedef __attribute__((ext_vector_type(16))) __bf16 v16bf;
typedef __attribute__((ext_vector_type(8)))  __bf16 v8bf;
typedef __attribute__((ext_vector_type(8)))  float  v8f;

#define TILE_M 256
#define TILE_N 128
#define TILE_K 32
#define LDSB   40  // bf16 elems per LDS row: 32 + 8 pad (80B rows; 16B-aligned chunks)

__device__ __forceinline__ __bf16 f2bf(float f) {
  union { float f; unsigned u; } c; c.f = f;
  unsigned u = c.u + 0x7FFFu + ((c.u >> 16) & 1u);  // RNE
  unsigned short h = (unsigned short)(u >> 16);
  __bf16 r;
  __builtin_memcpy(&r, &h, 2);
  return r;
}

__device__ __forceinline__ unsigned pack2bf(float x, float y) {
  union { float f; unsigned u; } a, b; a.f = x; b.f = y;
  unsigned ua = a.u + 0x7FFFu + ((a.u >> 16) & 1u);
  unsigned ub = b.u + 0x7FFFu + ((b.u >> 16) & 1u);
  return (ua >> 16) | (ub & 0xFFFF0000u);
}

// ---------------------------------------------------------------------------
// Uniform GEMM: C[m,n] = sum_k A[m,k] * B[n,k]   (A:[M,K] bf16, B:[N,K] bf16)
// OUT: 0 = f32 C[m*ldc+n] (+bias,+res)   1 = bf16 C[m*ldc+n] (+bias)
//      2 = bf16 transposed C[n*ldt+m] (+bias)  -- vectorized 16B stores
// M multiple of 256, N multiple of 128, K multiple of 32 (problem guarantees).
// ---------------------------------------------------------------------------
template <bool HAS_BIAS, bool HAS_RES, int OUT>
__global__ __launch_bounds__(256) void gemm_bf16_wmma(
    const __bf16* __restrict__ A, const __bf16* __restrict__ B,
    const float* __restrict__ bias, const float* __restrict__ res,
    float* __restrict__ Cf, __bf16* __restrict__ Cb,
    int K, int lda, int ldb, int ldc, int ldt) {
  __shared__ __align__(16) __bf16 As[2][TILE_M * LDSB];  // 2 x 20KB
  __shared__ __align__(16) __bf16 Bs[2][TILE_N * LDSB];  // 2 x 10KB

  const int t    = threadIdx.x;  // 256 threads = 8 waves
  const int wave = t >> 5;
  const int lane = t & 31;
  const int l    = lane & 15;
  const int hi   = lane >> 4;

  const int waveM = wave >> 1;  // 0..3 (64 rows each)
  const int waveN = wave & 1;   // 0..1 (64 cols each)

  const int blockM = blockIdx.y * TILE_M;
  const int blockN = blockIdx.x * TILE_N;

  v8f acc[4][4];
#pragma unroll
  for (int mi = 0; mi < 4; ++mi)
#pragma unroll
    for (int ni = 0; ni < 4; ++ni)
      acc[mi][ni] = (v8f){0.f, 0.f, 0.f, 0.f, 0.f, 0.f, 0.f, 0.f};

  // stage one k-tile (pure 16B copies, no conversion)
  auto stage = [&](int buf, int k0) {
#pragma unroll
    for (int i = 0; i < 4; ++i) {  // A: 256x32 bf16 = 1024 chunks
      const int f = i * 256 + t;
      const int m = f >> 2;
      const int c = (f & 3) << 3;
      *reinterpret_cast<uint4*>(&As[buf][m * LDSB + c]) =
          *reinterpret_cast<const uint4*>(&A[(size_t)(blockM + m) * lda + k0 + c]);
    }
#pragma unroll
    for (int i = 0; i < 2; ++i) {  // B: 128x32 bf16 = 512 chunks
      const int f = i * 256 + t;
      const int m = f >> 2;
      const int c = (f & 3) << 3;
      *reinterpret_cast<uint4*>(&Bs[buf][m * LDSB + c]) =
          *reinterpret_cast<const uint4*>(&B[(size_t)(blockN + m) * ldb + k0 + c]);
    }
  };

  stage(0, 0);
  __syncthreads();

  int buf = 0;
  for (int k0 = 0; k0 < K; k0 += TILE_K, buf ^= 1) {
    // stage next tile into the other buffer while computing this one
    if (k0 + TILE_K < K) stage(buf ^ 1, k0 + TILE_K);
    // prefetch two tiles ahead into near caches (global_prefetch_b8)
    if (k0 + 2 * TILE_K < K) {
      const int pr = t >> 1, ph = (t & 1) << 4;
      __builtin_prefetch(&A[(size_t)(blockM + pr) * lda + k0 + 2 * TILE_K + ph], 0, 3);
      __builtin_prefetch(&B[(size_t)(blockN + (pr & 127)) * ldb + k0 + 2 * TILE_K + ph], 0, 3);
    }

    // ---- B fragments (32x16): lane l -> N=l; elem e -> K = hi*16 + e ----
    v16bf bfr[4];
#pragma unroll
    for (int ni = 0; ni < 4; ++ni) {
      const __bf16* src = &Bs[buf][(waveN * 64 + ni * 16 + l) * LDSB + hi * 16];
      const v8bf lo = *reinterpret_cast<const v8bf*>(src);
      const v8bf up = *reinterpret_cast<const v8bf*>(src + 8);
      bfr[ni] = __builtin_shufflevector(lo, up, 0, 1, 2, 3, 4, 5, 6, 7, 8, 9,
                                        10, 11, 12, 13, 14, 15);
    }
    // ---- A fragments (16x32): lane l -> M=l; elem e -> K = hi*16 + e ----
#pragma unroll
    for (int mi = 0; mi < 4; ++mi) {
      const __bf16* src = &As[buf][(waveM * 64 + mi * 16 + l) * LDSB + hi * 16];
      const v8bf lo = *reinterpret_cast<const v8bf*>(src);
      const v8bf up = *reinterpret_cast<const v8bf*>(src + 8);
      const v16bf afr = __builtin_shufflevector(lo, up, 0, 1, 2, 3, 4, 5, 6, 7,
                                                8, 9, 10, 11, 12, 13, 14, 15);
#pragma unroll
      for (int ni = 0; ni < 4; ++ni) {
        acc[mi][ni] = __builtin_amdgcn_wmma_f32_16x16x32_bf16(
            false, afr, false, bfr[ni], (short)0, acc[mi][ni], false, false);
      }
    }
    __syncthreads();  // next-tile stores complete + this-tile reads done
  }

  // ---- epilogue: D elem r -> row M = m0 + r (m0 incl. hi*8), col N = l ----
#pragma unroll
  for (int mi = 0; mi < 4; ++mi) {
#pragma unroll
    for (int ni = 0; ni < 4; ++ni) {
      const int n  = blockN + waveN * 64 + ni * 16 + l;
      const int m0 = blockM + waveM * 64 + mi * 16 + hi * 8;
      const float b = HAS_BIAS ? bias[n] : 0.f;
      if (OUT == 2) {
        v8bf o;
#pragma unroll
        for (int r = 0; r < 8; ++r) o[r] = f2bf(acc[mi][ni][r] + b);
        *reinterpret_cast<v8bf*>(&Cb[(size_t)n * ldt + m0]) = o;  // 16B store
      } else {
#pragma unroll
        for (int r = 0; r < 8; ++r) {
          float v = acc[mi][ni][r] + b;
          if (HAS_RES) v += res[(size_t)(m0 + r) * ldc + n];
          if (OUT == 0) Cf[(size_t)(m0 + r) * ldc + n] = v;
          else          Cb[(size_t)(m0 + r) * ldc + n] = f2bf(v);
        }
      }
    }
  }
}

// ---------------------------------------------------------------------------
// fp32 -> bf16 elementwise (8 elems / thread, 32B in / 16B out)
__global__ __launch_bounds__(256) void cvt_f32_bf16(const float* __restrict__ in,
                                                    __bf16* __restrict__ out) {
  const size_t i = (size_t)blockIdx.x * 256 + threadIdx.x;
  const float4 a = reinterpret_cast<const float4*>(in)[2 * i];
  const float4 b = reinterpret_cast<const float4*>(in)[2 * i + 1];
  uint4 o;
  o.x = pack2bf(a.x, a.y);
  o.y = pack2bf(a.z, a.w);
  o.z = pack2bf(b.x, b.y);
  o.w = pack2bf(b.z, b.w);
  reinterpret_cast<uint4*>(out)[i] = o;
}

// fp32 [R,C] -> bf16 transposed [C,R]:  Wt[o*R + i] = W[i*C + o]
__global__ __launch_bounds__(256) void transpose_cvt_bf16(
    const float* __restrict__ W, __bf16* __restrict__ Wt, int R, int C) {
  __shared__ float tile[32][33];
  const int r = threadIdx.x >> 5;   // 0..7
  const int c = threadIdx.x & 31;   // 0..31
  const int bi = blockIdx.y * 32;   // input row base
  const int bo = blockIdx.x * 32;   // input col base
#pragma unroll
  for (int p = 0; p < 4; ++p)
    tile[r + p * 8][c] = W[(size_t)(bi + r + p * 8) * C + bo + c];
  __syncthreads();
#pragma unroll
  for (int p = 0; p < 4; ++p)
    Wt[(size_t)(bo + r + p * 8) * R + bi + c] = f2bf(tile[c][r + p * 8]);
}

// Row softmax over 2048 cols, in place (fp32), plus bf16 copy for the next GEMM.
__global__ __launch_bounds__(256) void softmax2048(float* __restrict__ data,
                                                   __bf16* __restrict__ datab) {
  float* p = data + (size_t)blockIdx.x * 2048;
  __bf16* pb = datab + (size_t)blockIdx.x * 2048;
  const int t = threadIdx.x;
  __shared__ float s[256];

  float v[8];
  float m = -INFINITY;
#pragma unroll
  for (int i = 0; i < 8; ++i) {
    v[i] = p[t + i * 256];
    m = fmaxf(m, v[i]);
  }
  s[t] = m;
  __syncthreads();
  for (int off = 128; off > 0; off >>= 1) {
    if (t < off) s[t] = fmaxf(s[t], s[t + off]);
    __syncthreads();
  }
  m = s[0];
  __syncthreads();

  float sum = 0.f;
#pragma unroll
  for (int i = 0; i < 8; ++i) {
    v[i] = __expf(v[i] - m);
    sum += v[i];
  }
  s[t] = sum;
  __syncthreads();
  for (int off = 128; off > 0; off >>= 1) {
    if (t < off) s[t] += s[t + off];
    __syncthreads();
  }
  const float inv = 1.0f / s[0];
#pragma unroll
  for (int i = 0; i < 8; ++i) {
    const float w = v[i] * inv;
    p[t + i * 256]  = w;
    pb[t + i * 256] = f2bf(w);
  }
}

// ---------------------------------------------------------------------------

extern "C" void kernel_launch(void* const* d_in, const int* in_sizes, int n_in,
                              void* d_out, int out_size, void* d_ws, size_t ws_size,
                              hipStream_t stream) {
  (void)in_sizes; (void)n_in; (void)out_size; (void)ws_size;

  const int N = 8192, D = 2048;
  const size_t NM = (size_t)N * D;
  const size_t DD = (size_t)D * D;

  const float* x = (const float*)d_in[0];
  const float* y = (const float*)d_in[1];
  const float* W[8];   // QK,QQ,QV,QO, FK,FQ,FV,FO weights
  const float* Bv[8];  // matching biases
  for (int i = 0; i < 4; ++i) { W[i]      = (const float*)d_in[2 + 2 * i];
                                Bv[i]     = (const float*)d_in[3 + 2 * i]; }
  for (int i = 0; i < 4; ++i) { W[4 + i]  = (const float*)d_in[10 + 2 * i];
                                Bv[4 + i] = (const float*)d_in[11 + 2 * i]; }

  float* out      = (float*)d_out;
  float* outQ_y   = out;
  float* outQ_att = out + NM;
  float* outF_y   = out + NM + DD;
  float* outF_att = out + 2 * NM + DD;

  // workspace (bf16), reused across branches: 4*DD + 4*NM + DD elems (~176MB)
  __bf16* WtK = (__bf16*)d_ws;      // [D,D] transposed weights
  __bf16* WtQ = WtK + DD;
  __bf16* WtV = WtQ + DD;
  __bf16* WtO = WtV + DD;
  __bf16* xbf = WtO + DD;           // [N,D]; reused as V*w^T temp
  __bf16* Qt  = xbf + NM;           // [D,N]
  __bf16* Kt  = Qt + NM;            // [D,N]
  __bf16* Vb  = Kt + NM;            // [N,D]
  __bf16* wb  = Vb + NM;            // [D,D]

  const dim3 blk(256);
  const dim3 gNM(D / TILE_N, N / TILE_M);   // 16 x 32
  const dim3 gDD(D / TILE_N, D / TILE_M);   // 16 x 8
  const dim3 tgrid(D / 32, D / 32);
  const int  cvtBlocksNM = (int)(NM / (8 * 256));

  auto run_branch = [&](const float* inp, const float* Kw, const float* Kb,
                        const float* Qw, const float* Qb, const float* Vw,
                        const float* Vb_, const float* Ow, const float* Ob,
                        float* out_y, float* out_att) {
    // one-time conversions (memory-bound, tiny vs GEMM work)
    cvt_f32_bf16<<<cvtBlocksNM, blk, 0, stream>>>(inp, xbf);
    transpose_cvt_bf16<<<tgrid, blk, 0, stream>>>(Kw, WtK, D, D);
    transpose_cvt_bf16<<<tgrid, blk, 0, stream>>>(Qw, WtQ, D, D);
    transpose_cvt_bf16<<<tgrid, blk, 0, stream>>>(Vw, WtV, D, D);
    transpose_cvt_bf16<<<tgrid, blk, 0, stream>>>(Ow, WtO, D, D);

    // K/Q projections -> transposed bf16 [D,N]; V -> bf16 [N,D]
    gemm_bf16_wmma<true, false, 2><<<gNM, blk, 0, stream>>>(
        xbf, WtK, Kb, nullptr, nullptr, Kt, D, D, D, 0, N);
    gemm_bf16_wmma<true, false, 2><<<gNM, blk, 0, stream>>>(
        xbf, WtQ, Qb, nullptr, nullptr, Qt, D, D, D, 0, N);
    gemm_bf16_wmma<true, false, 1><<<gNM, blk, 0, stream>>>(
        xbf, WtV, Vb_, nullptr, nullptr, Vb, D, D, D, D, 0);
    // att[i,j] = sum_n Qt[i,n]*Kt[j,n]  -> fp32 att slot of d_out
    gemm_bf16_wmma<false, false, 0><<<gDD, blk, 0, stream>>>(
        Qt, Kt, nullptr, nullptr, out_att, nullptr, N, N, N, D, 0);
    // w = softmax(att) in place + bf16 copy
    softmax2048<<<D, blk, 0, stream>>>(out_att, wb);
    // tmp[n,i] = sum_j Vb[n,j]*wb[i,j] -> bf16 (reuse xbf)
    gemm_bf16_wmma<false, false, 1><<<gNM, blk, 0, stream>>>(
        Vb, wb, nullptr, nullptr, nullptr, xbf, D, D, D, D, 0);
    // y = tmp @ Ow + Ob + inp -> fp32
    gemm_bf16_wmma<true, true, 0><<<gNM, blk, 0, stream>>>(
        xbf, WtO, Ob, inp, out_y, nullptr, D, D, D, D, 0);
  };

  run_branch(x, W[0], Bv[0], W[1], Bv[1], W[2], Bv[2], W[3], Bv[3], outQ_y, outQ_att);
  run_branch(y, W[4], Bv[4], W[5], Bv[5], W[6], Bv[6], W[7], Bv[7], outF_y, outF_att);
}
